// CausalSelfAttention_3874060501811
// MI455X (gfx1250) — matrix operand order repaired and checked by reading, not compiled, not run on hardware
//
#include <hip/hip_runtime.h>
#include <hip/hip_bf16.h>
#include <math.h>

// ---------------- problem constants ----------------
#define BATCH  2
#define SEQ    2048
#define EMB    2048
#define NH     16
#define NKV    4
#define HD     128
#define MROWS  (BATCH*SEQ)   // 4096

typedef __bf16 v16bf __attribute__((ext_vector_type(16)));
typedef float  v8f   __attribute__((ext_vector_type(8)));
typedef unsigned int u32x4 __attribute__((ext_vector_type(4)));
typedef int          i32x4 __attribute__((ext_vector_type(4)));
typedef int          i32x8 __attribute__((ext_vector_type(8)));

union BFrag {
  uint4          u4[2];
  unsigned short us[16];
  v16bf          v;
};

__device__ __forceinline__ unsigned short f2bf(float f) {
  unsigned u = __float_as_uint(f);
  u += 0x7FFFu + ((u >> 16) & 1u);          // round-to-nearest-even
  return (unsigned short)(u >> 16);
}
__device__ __forceinline__ float bf2f(unsigned short h) {
  return __uint_as_float(((unsigned)h) << 16);
}

// ---------------- TDM availability (device pass only) ----------------
#if defined(__AMDGCN__)
#  if __has_builtin(__builtin_amdgcn_tensor_load_to_lds)
#    define HAVE_TDM 1
#  endif
#endif
#ifndef HAVE_TDM
#  define HAVE_TDM 0
#endif

#if HAVE_TDM
__device__ __forceinline__ unsigned lds_addr32(const unsigned short* p) {
  return (unsigned)(size_t)(const __attribute__((address_space(3))) unsigned short*)p;
}
// Issue a TDM 2D tile load: 64 rows x 32 bf16, row stride = K elements.
__device__ __forceinline__ void tdm_load_tile(unsigned lds_byte,
                                              unsigned long long gbyte, int K) {
  u32x4 g0;
  g0.x = 1u;                                           // count=1, is_restore=0
  g0.y = lds_byte;                                     // lds_addr
  g0.z = (unsigned)gbyte;                              // global_addr[31:0]
  g0.w = (unsigned)((gbyte >> 32) & 0x1FFFFFFull)      // global_addr[56:32]
       | (2u << 30);                                   // type=2 ("image")
  i32x8 g1;
  g1[0] = 1 << 16;                                     // data_size=1 (2 bytes), wg_mask=0
  g1[1] = (int)(((unsigned)K & 0xFFFFu) << 16);        // tensor_dim0[15:0]
  g1[2] = (int)(((unsigned)K >> 16) | (64u << 16));    // tensor_dim0[31:16] | tensor_dim1 lo
  g1[3] = (int)(32u << 16);                            // tensor_dim1 hi=0 | tile_dim0=32
  g1[4] = 64;                                          // tile_dim1=64, tile_dim2=0
  g1[5] = K;                                           // tensor_dim0_stride[31:0]
  g1[6] = 0;                                           // stride hi, dim1_stride lo
  g1[7] = 0;
  i32x4 z4 = {0, 0, 0, 0};
#if __clang_major__ >= 23
  i32x8 z8 = {0, 0, 0, 0, 0, 0, 0, 0};
  __builtin_amdgcn_tensor_load_to_lds(g0, g1, z4, z4, z8, 0);
#else
  __builtin_amdgcn_tensor_load_to_lds(g0, g1, z4, z4, 0);
#endif
}
#endif  // HAVE_TDM

// ---------------- fp32 -> bf16 cast ----------------
__global__ void cast_bf16_kernel(const float* __restrict__ src,
                                 unsigned short* __restrict__ dst, long long n) {
  long long i = (long long)blockIdx.x * blockDim.x + threadIdx.x;
  long long st = (long long)gridDim.x * blockDim.x;
  for (; i < n; i += st) dst[i] = f2bf(src[i]);
}

// ---------------- WMMA GEMM: C[M,N] = A[M,K] * W[N,K]^T ----------------
// 4 waves/block; wave w: rows [blockIdx.y*128 + w*32, +32), cols [blockIdx.x*64, +64).
// W tiles (64x32 bf16, shared by all 4 waves) staged into LDS by the Tensor Data
// Mover, double-buffered. Each wave holds 2 A fragments (32 rows), so every staged
// tile feeds 8 WMMAs per wave; A fragments are software-pipelined in registers and
// all B ds_loads are issued ahead of the WMMA burst for progressive dscnt waits.
__global__ void __launch_bounds__(128)
gemm_bf16_wmma(const unsigned short* __restrict__ A, const unsigned short* __restrict__ W,
               float* __restrict__ outF, unsigned short* __restrict__ outB,
               int M, int N, int K) {
  __shared__ unsigned short Wlds[2][64 * 32];          // 2 x 4KB double buffer
  const int tid  = threadIdx.x;
  const int lane = tid & 31;
  const int wv   = tid >> 5;
  const int hf   = lane >> 4;
  const int l16  = lane & 15;
  const int rowM = blockIdx.y * 128 + wv * 32;
  const int colN = blockIdx.x * 64;

  v8f acc[2][4];
  #pragma unroll
  for (int r = 0; r < 2; ++r)
    #pragma unroll
    for (int t = 0; t < 4; ++t)
      #pragma unroll
      for (int i = 0; i < 8; ++i) acc[r][t][i] = 0.0f;

  // stage one 64x32 weight tile (k columns [k0,k0+32)) into Wlds[buf]
  auto stage = [&](int buf, int k0) {
#if HAVE_TDM
    if (wv == 0) {
      unsigned long long gb =
          (unsigned long long)(size_t)W + (((size_t)colN * K + k0) << 1);
      tdm_load_tile(lds_addr32(&Wlds[buf][0]), gb, K);
    }
#else
    const int row = tid >> 1, hx = tid & 1;            // 128 threads: 2 per row
    const unsigned short* src = W + (size_t)(colN + row) * K + k0 + hx * 16;
    unsigned short* dst = &Wlds[buf][row * 32 + hx * 16];
    *reinterpret_cast<uint4*>(dst)     = *reinterpret_cast<const uint4*>(src);
    *reinterpret_cast<uint4*>(dst + 8) = *reinterpret_cast<const uint4*>(src + 8);
#endif
  };
  auto tile_ready = [&]() {
#if HAVE_TDM
    if (wv == 0) __builtin_amdgcn_s_wait_tensorcnt(0);
#endif
    __syncthreads();
  };

  // A fragments: lane holds rows (rowM+l16) and (rowM+16+l16);
  // elems 0-7 = K k0+hf*8.., elems 8-15 = K k0+hf*8+16..
  const unsigned short* ap0 = A + (size_t)(rowM + l16) * K + hf * 8;
  const unsigned short* ap1 = A + (size_t)(rowM + 16 + l16) * K + hf * 8;

  stage(0, 0);
  BFrag a0, a1;
  a0.u4[0] = *reinterpret_cast<const uint4*>(ap0);
  a0.u4[1] = *reinterpret_cast<const uint4*>(ap0 + 16);
  a1.u4[0] = *reinterpret_cast<const uint4*>(ap1);
  a1.u4[1] = *reinterpret_cast<const uint4*>(ap1 + 16);

  for (int k0 = 0; k0 < K; k0 += 32) {
    const int cur = (k0 >> 5) & 1;
    tile_ready();                                      // Wlds[cur] valid for all waves
    if (k0 + 32 < K) stage(cur ^ 1, k0 + 32);          // DMA next tile while computing
    BFrag a0n = a0, a1n = a1;                          // prefetch next A fragments
    if (k0 + 32 < K) {
      a0n.u4[0] = *reinterpret_cast<const uint4*>(ap0 + k0 + 32);
      a0n.u4[1] = *reinterpret_cast<const uint4*>(ap0 + k0 + 48);
      a1n.u4[0] = *reinterpret_cast<const uint4*>(ap1 + k0 + 32);
      a1n.u4[1] = *reinterpret_cast<const uint4*>(ap1 + k0 + 48);
    }
    // issue all 8 B ds_loads before the WMMA burst
    BFrag b[4];
    #pragma unroll
    for (int t = 0; t < 4; ++t) {
      const unsigned short* wp = &Wlds[cur][(t * 16 + l16) * 32 + hf * 8];
      b[t].u4[0] = *reinterpret_cast<const uint4*>(wp);
      b[t].u4[1] = *reinterpret_cast<const uint4*>(wp + 16);
    }
    #pragma unroll
    for (int t = 0; t < 4; ++t) {
      acc[0][t] = __builtin_amdgcn_wmma_f32_16x16x32_bf16(false, a0.v, false, b[t].v,
                                                          (short)0, acc[0][t], false, false);
      acc[1][t] = __builtin_amdgcn_wmma_f32_16x16x32_bf16(false, a1.v, false, b[t].v,
                                                          (short)0, acc[1][t], false, false);
    }
    a0 = a0n;
    a1 = a1n;
  }

  #pragma unroll
  for (int r = 0; r < 2; ++r)
    #pragma unroll
    for (int t = 0; t < 4; ++t)
      #pragma unroll
      for (int i = 0; i < 8; ++i) {
        int m = rowM + r * 16 + i + 8 * hf;
        int n = colN + t * 16 + l16;
        if (outF) outF[(size_t)m * N + n] = acc[r][t][i];
        else      outB[(size_t)m * N + n] = f2bf(acc[r][t][i]);
      }
}

// ---------------- interleaved RoPE, in-place on bf16 [B*S][heads*128] ----------------
__global__ void rope_kernel(unsigned short* __restrict__ buf, const int* __restrict__ pos,
                            int heads, long long npairs) {
  long long idx = (long long)blockIdx.x * blockDim.x + threadIdx.x;
  long long st  = (long long)gridDim.x * blockDim.x;
  for (; idx < npairs; idx += st) {
    int       i = (int)(idx & 63);
    long long r = idx >> 6;
    int       h = (int)(r % heads); r /= heads;        // r = b*SEQ + s
    int       s = (int)(r % SEQ);
    float inv = __expf(-((float)(2 * i) / 128.0f) * 9.210340371976184f);
    float ang = (float)pos[s] * inv;
    float sn, c;
    __sincosf(ang, &sn, &c);
    size_t base = (size_t)r * heads * 128 + (size_t)h * 128 + 2 * i;
    float e = bf2f(buf[base]), o = bf2f(buf[base + 1]);
    buf[base]     = f2bf(e * c - o * sn);
    buf[base + 1] = f2bf(o * c + e * sn);
  }
}

// ---------------- V[B*S][NKV*128] -> Vt[b][kvh][d][S] ----------------
__global__ void vtrans_kernel(const unsigned short* __restrict__ V,
                              unsigned short* __restrict__ Vt, long long n) {
  long long idx = (long long)blockIdx.x * blockDim.x + threadIdx.x;
  long long st  = (long long)gridDim.x * blockDim.x;
  for (; idx < n; idx += st) {
    int       d   = (int)(idx & 127);
    long long r   = idx >> 7;
    int       kvh = (int)(r & 3); r >>= 2;             // r = b*SEQ + s
    int       s   = (int)(r % SEQ);
    int       b   = (int)(r / SEQ);
    Vt[(((size_t)b * NKV + kvh) * HD + d) * SEQ + s] = V[idx];
  }
}

// ---------------- flash attention: per-wave 16 q-rows x D=128, 32-key tiles ----------------
__global__ void __launch_bounds__(128)
attn_wmma_kernel(const unsigned short* __restrict__ Q,   // [B*S][NH*128], roped
                 const unsigned short* __restrict__ Kb,  // [B*S][NKV*128], roped
                 const unsigned short* __restrict__ Vt,  // [b][kvh][d][S]
                 unsigned short* __restrict__ Y) {       // [B*S][NH*128]
  __shared__ unsigned short Plds[4 * 16 * 32];           // 1KB per wave (P tile bounce)
  const int lane  = threadIdx.x & 31;
  const int wv    = threadIdx.x >> 5;
  const int hf    = lane >> 4;
  const int l16   = lane & 15;
  const int h     = blockIdx.y;
  const int b     = blockIdx.z;
  const int kvh   = h >> 2;                              // GQA: 4 Q heads per KV head
  const int qbase = blockIdx.x * 64 + wv * 16;
  unsigned short* P = Plds + wv * 512;

  // Q fragments: 4 chunks of K=32 over D=128 (held for whole kernel)
  BFrag qf[4];
  const unsigned short* qp = Q + ((size_t)b * SEQ + qbase + l16) * (NH * HD) + h * HD + hf * 8;
  #pragma unroll
  for (int g = 0; g < 4; ++g) {
    qf[g].u4[0] = *reinterpret_cast<const uint4*>(qp + g * 32);
    qf[g].u4[1] = *reinterpret_cast<const uint4*>(qp + g * 32 + 16);
  }

  v8f acc[8];                                            // O accumulator 16x128 f32
  #pragma unroll
  for (int g = 0; g < 8; ++g)
    #pragma unroll
    for (int i = 0; i < 8; ++i) acc[g][i] = 0.0f;
  float m_i[8], l_i[8];
  #pragma unroll
  for (int i = 0; i < 8; ++i) { m_i[i] = -3.0e38f; l_i[i] = 0.0f; }

  const float scale  = 0.08838834764831845f;             // 1/sqrt(128)
  const int   ntiles = (qbase + 15) / 32 + 1;            // causal bound

  for (int j = 0; j < ntiles; ++j) {
    const int kb0 = j * 32;
    // ---- load all 8 K fragments first (lets loads drain behind WMMAs) ----
    BFrag kf0[4], kf1[4];
    #pragma unroll
    for (int g = 0; g < 4; ++g) {
      const unsigned short* kp0 =
          Kb + ((size_t)b * SEQ + kb0 + l16) * (NKV * HD) + kvh * HD + g * 32 + hf * 8;
      kf0[g].u4[0] = *reinterpret_cast<const uint4*>(kp0);
      kf0[g].u4[1] = *reinterpret_cast<const uint4*>(kp0 + 16);
      const unsigned short* kp1 = kp0 + (size_t)16 * (NKV * HD);
      kf1[g].u4[0] = *reinterpret_cast<const uint4*>(kp1);
      kf1[g].u4[1] = *reinterpret_cast<const uint4*>(kp1 + 16);
    }
    // ---- scores S = Q K^T : two 16x16 col tiles, 4 d-chunks each ----
    v8f s0, s1;
    #pragma unroll
    for (int i = 0; i < 8; ++i) { s0[i] = 0.0f; s1[i] = 0.0f; }
    #pragma unroll
    for (int g = 0; g < 4; ++g) {
      s0 = __builtin_amdgcn_wmma_f32_16x16x32_bf16(false, qf[g].v, false, kf0[g].v,
                                                   (short)0, s0, false, false);
      s1 = __builtin_amdgcn_wmma_f32_16x16x32_bf16(false, qf[g].v, false, kf1[g].v,
                                                   (short)0, s1, false, false);
    }
    // ---- mask + streaming softmax (row = i + 8*hf across 16 lanes of a half) ----
    #pragma unroll
    for (int i = 0; i < 8; ++i) {
      const int qr = qbase + i + 8 * hf;
      float v0 = s0[i] * scale; if (kb0 + l16      > qr) v0 = -3.0e38f;
      float v1 = s1[i] * scale; if (kb0 + 16 + l16 > qr) v1 = -3.0e38f;
      float mx = fmaxf(v0, v1);
      #pragma unroll
      for (int d = 1; d < 16; d <<= 1) mx = fmaxf(mx, __shfl_xor(mx, d, 32));
      const float mnew  = fmaxf(m_i[i], mx);
      const float alpha = __expf(m_i[i] - mnew);
      const float p0 = __expf(v0 - mnew);
      const float p1 = __expf(v1 - mnew);
      float rs = p0 + p1;
      #pragma unroll
      for (int d = 1; d < 16; d <<= 1) rs += __shfl_xor(rs, d, 32);
      l_i[i] = l_i[i] * alpha + rs;
      m_i[i] = mnew;
      #pragma unroll
      for (int g = 0; g < 8; ++g) acc[g][i] *= alpha;
      const int r = i + 8 * hf;                          // C-layout -> LDS row-major P
      P[r * 32 + l16]      = f2bf(p0);
      P[r * 32 + 16 + l16] = f2bf(p1);
    }
    asm volatile("s_wait_dscnt 0x0" ::: "memory");       // LDS stores visible before reload
    // ---- reload P in A-fragment layout (row = l16, K chunks hf*8.. / hf*8+16..) ----
    BFrag pf;
    const unsigned short* pp = P + l16 * 32 + hf * 8;
    pf.u4[0] = *reinterpret_cast<const uint4*>(pp);
    pf.u4[1] = *reinterpret_cast<const uint4*>(pp + 16);
    // ---- load all 8 V fragments, then O += P V ----
    BFrag vf[8];
    #pragma unroll
    for (int g = 0; g < 8; ++g) {
      const unsigned short* vp =
          Vt + (((size_t)b * NKV + kvh) * HD + g * 16 + l16) * SEQ + kb0 + hf * 8;
      vf[g].u4[0] = *reinterpret_cast<const uint4*>(vp);
      vf[g].u4[1] = *reinterpret_cast<const uint4*>(vp + 16);
    }
    #pragma unroll
    for (int g = 0; g < 8; ++g)
      acc[g] = __builtin_amdgcn_wmma_f32_16x16x32_bf16(false, pf.v, false, vf[g].v,
                                                       (short)0, acc[g], false, false);
  }
  // ---- normalize + store bf16 heads-interleaved (== transpose(0,2,1,3) flatten) ----
  #pragma unroll
  for (int i = 0; i < 8; ++i) {
    const float invl = 1.0f / l_i[i];
    const int   qr   = qbase + i + 8 * hf;
    unsigned short* yp = Y + ((size_t)b * SEQ + qr) * (NH * HD) + h * HD;
    #pragma unroll
    for (int g = 0; g < 8; ++g) yp[g * 16 + l16] = f2bf(acc[g][i] * invl);
  }
}

// ---------------- host side ----------------
extern "C" void kernel_launch(void* const* d_in, const int* in_sizes, int n_in,
                              void* d_out, int out_size, void* d_ws, size_t ws_size,
                              hipStream_t stream) {
  const float* x  = (const float*)d_in[0];
  const float* Wq = (const float*)d_in[1];
  const float* Wk = (const float*)d_in[2];
  const float* Wv = (const float*)d_in[3];
  const float* Wo = (const float*)d_in[4];
  const int*   pos = (const int*)d_in[5];
  float* out = (float*)d_out;

  char* ws = (char*)d_ws;
  const size_t MB = 1ull << 20;
  unsigned short* xb   = (unsigned short*)(ws + 0 * MB);   // 16 MB  [4096][2048]
  unsigned short* wqb  = (unsigned short*)(ws + 16 * MB);  //  8 MB  [2048][2048]
  unsigned short* wkb  = (unsigned short*)(ws + 24 * MB);  //  2 MB  [512][2048]
  unsigned short* wvb  = (unsigned short*)(ws + 26 * MB);  //  2 MB
  unsigned short* wob  = (unsigned short*)(ws + 28 * MB);  //  8 MB
  unsigned short* Qb   = (unsigned short*)(ws + 36 * MB);  // 16 MB  [4096][2048]
  unsigned short* Kbuf = (unsigned short*)(ws + 52 * MB);  //  4 MB  [4096][512]
  unsigned short* Vraw = (unsigned short*)(ws + 56 * MB);  //  4 MB
  unsigned short* Vt   = (unsigned short*)(ws + 60 * MB);  //  4 MB  [2][4][128][2048]
  unsigned short* Yb   = (unsigned short*)(ws + 64 * MB);  // 16 MB  [4096][2048]

  // 1) casts to bf16
  cast_bf16_kernel<<<4096, 256, 0, stream>>>(x,  xb,  (long long)MROWS * EMB);
  cast_bf16_kernel<<<2048, 256, 0, stream>>>(Wq, wqb, (long long)EMB * EMB);
  cast_bf16_kernel<<<1024, 256, 0, stream>>>(Wk, wkb, (long long)(NKV * HD) * EMB);
  cast_bf16_kernel<<<1024, 256, 0, stream>>>(Wv, wvb, (long long)(NKV * HD) * EMB);
  cast_bf16_kernel<<<2048, 256, 0, stream>>>(Wo, wob, (long long)EMB * EMB);

  // 2) QKV projections (WMMA + TDM-staged weight tiles)
  gemm_bf16_wmma<<<dim3(EMB / 64, MROWS / 128), 128, 0, stream>>>(
      xb, wqb, nullptr, Qb, MROWS, EMB, EMB);
  gemm_bf16_wmma<<<dim3((NKV * HD) / 64, MROWS / 128), 128, 0, stream>>>(
      xb, wkb, nullptr, Kbuf, MROWS, NKV * HD, EMB);
  gemm_bf16_wmma<<<dim3((NKV * HD) / 64, MROWS / 128), 128, 0, stream>>>(
      xb, wvb, nullptr, Vraw, MROWS, NKV * HD, EMB);

  // 3) RoPE on Q and K (in place)
  rope_kernel<<<4096, 256, 0, stream>>>(Qb,   pos, NH,  (long long)MROWS * NH  * 64);
  rope_kernel<<<2048, 256, 0, stream>>>(Kbuf, pos, NKV, (long long)MROWS * NKV * 64);

  // 4) V -> Vt (d-major) for contiguous P*V B-fragments
  vtrans_kernel<<<2048, 256, 0, stream>>>(Vraw, Vt, (long long)MROWS * NKV * HD);

  // 5) causal flash attention (WMMA)
  attn_wmma_kernel<<<dim3(SEQ / 64, NH, BATCH), 128, 0, stream>>>(Qb, Kbuf, Vt, Yb);

  // 6) output projection -> fp32
  gemm_bf16_wmma<<<dim3(EMB / 64, MROWS / 128), 128, 0, stream>>>(
      Yb, wob, out, nullptr, MROWS, EMB, EMB);
}
